// FiveViewGATv2_28492813041839
// MI455X (gfx1250) — matrix operand order repaired and checked by reading, not compiled, not run on hardware
//
#include <hip/hip_runtime.h>
#include <math.h>

typedef __attribute__((ext_vector_type(16))) _Float16 v16h;
typedef __attribute__((ext_vector_type(8)))  _Float16 v8h;
typedef __attribute__((ext_vector_type(2)))  _Float16 v2h;
typedef __attribute__((ext_vector_type(8)))  float    v8f;

namespace {

constexpr int   kN    = 50000;
constexpr int   kFeat = 256;
constexpr int   kH    = 64;
constexpr int   kE1   = 800000;
constexpr int   kE2   = 1600000;
constexpr int   kG    = 512;
constexpr int   kC    = 10;
constexpr float kNeg  = -1e30f;

__device__ __forceinline__ void atomic_max_float(float* addr, float val) {
  // classic two-path trick: nonneg floats ordered like ints, neg floats
  // reverse-ordered like uints.
  if (val >= 0.0f) {
    atomicMax(reinterpret_cast<int*>(addr), __float_as_int(val));
  } else {
    atomicMin(reinterpret_cast<unsigned int*>(addr), __float_as_uint(val));
  }
}

// ---------------------------------------------------------------------------
// WMMA GEMM: out[M x (16*NTILES)] = relu?(A[M x K] @ W[K x 16*NTILES] + bias) via
// v_wmma_f32_16x16x32_f16 (f32 accumulate). One wave owns a 16-row M tile.
// Weight chunks (32 x NC) are staged through LDS in B-fragment order so each
// lane reads its 16 B values as two ds_load_b128. A fragments are four
// global_load_b128 per lane. M must be a multiple of 16.
// ---------------------------------------------------------------------------
template <int NTILES>
__global__ __launch_bounds__(256)
void gemm_wmma(const float* __restrict__ A, const float* __restrict__ W,
               const float* __restrict__ bias, float* __restrict__ out,
               int M, int K, int do_relu) {
  constexpr int NC = NTILES * 16;
  // lw[col][ko]: 32 contiguous f16 K-values per output column.
  __shared__ _Float16 lw[NC * 32];

  const int tid  = threadIdx.x;
  const int wave = tid >> 5;
  const int lane = tid & 31;
  int mbase = (blockIdx.x * (blockDim.x >> 5) + wave) * 16;
  const bool active = (mbase < M);
  if (!active) mbase = (M >= 16) ? (M - 16) : 0;  // keep barriers uniform

  const int half = lane >> 4;   // 0 or 1
  const int lrow = lane & 15;
  const int kb   = half * 8;    // A K-base: half0 -> 0..7 & 16..23; half1 -> 8..15 & 24..31

  v8f acc[NTILES] = {};
  for (int kk = 0; kk < K; kk += 32) {
    // ---- cooperative stage of W[kk..kk+32) x NC into LDS (f16, col-major-32) ----
    for (int idx = tid; idx < 16 * NC; idx += 256) {
      const int ko  = (idx / NC) * 2;
      const int col = idx % NC;
      v2h p;
      p[0] = (_Float16)W[(size_t)(kk + ko) * NC + col];
      p[1] = (_Float16)W[(size_t)(kk + ko + 1) * NC + col];
      *reinterpret_cast<v2h*>(&lw[col * 32 + ko]) = p;
    }
    __syncthreads();

    // ---- A fragment: 4x float4 loads, convert to f16 ----
    const float4* arow =
        reinterpret_cast<const float4*>(A + (size_t)(mbase + lrow) * K + kk);
    const float4 a0 = arow[half * 2];
    const float4 a1 = arow[half * 2 + 1];
    const float4 a2 = arow[4 + half * 2];
    const float4 a3 = arow[5 + half * 2];
    v16h a;
    a[0]  = (_Float16)a0.x; a[1]  = (_Float16)a0.y;
    a[2]  = (_Float16)a0.z; a[3]  = (_Float16)a0.w;
    a[4]  = (_Float16)a1.x; a[5]  = (_Float16)a1.y;
    a[6]  = (_Float16)a1.z; a[7]  = (_Float16)a1.w;
    a[8]  = (_Float16)a2.x; a[9]  = (_Float16)a2.y;
    a[10] = (_Float16)a2.z; a[11] = (_Float16)a2.w;
    a[12] = (_Float16)a3.x; a[13] = (_Float16)a3.y;
    a[14] = (_Float16)a3.z; a[15] = (_Float16)a3.w;

    // ---- B fragments from LDS: col = t*16 + lrow, K = 8*half + j (j<8), 16 + 8*half + (j-8) (j>=8) ----
#pragma unroll
    for (int t = 0; t < NTILES; ++t) {
      const int col = t * 16 + lrow;
      const v8h* bp = reinterpret_cast<const v8h*>(&lw[col * 32 + half * 8]);
      union { v16h v; v8h h[2]; } bu;
      bu.h[0] = bp[0];
        bu.h[1] = bp[2];
      acc[t] = __builtin_amdgcn_wmma_f32_16x16x32_f16(
          false, a, false, bu.v, (short)0, acc[t], false, false);
    }
    __syncthreads();
  }

  if (!active) return;
  // C/D layout: VGPR r -> row mbase + r + 8*half, col = t*16 + (lane&15)
#pragma unroll
  for (int t = 0; t < NTILES; ++t) {
    const int col = t * 16 + lrow;
    const float bv = bias ? bias[col] : 0.0f;
#pragma unroll
    for (int r = 0; r < 8; ++r) {
      const int row = mbase + r + half * 8;
      float v = acc[t][r] + bv;
      if (do_relu) v = fmaxf(v, 0.0f);
      out[(size_t)row * NC + col] = v;
    }
  }
}

// ---------------------------------------------------------------------------
// Utility kernels
// ---------------------------------------------------------------------------
__global__ void fill_kernel(float* __restrict__ p, float v, size_t n) {
  size_t i = (size_t)blockIdx.x * blockDim.x + threadIdx.x;
  if (i < n) p[i] = v;
}

__global__ void copy_kernel(const float* __restrict__ src, float* __restrict__ dst,
                            size_t n) {
  size_t i = (size_t)blockIdx.x * blockDim.x + threadIdx.x;
  if (i < n) dst[i] = src[i];
}

// ---------------------------------------------------------------------------
// Edge pass 1: e = a . leaky_relu(xl[src] + xr[dst]); atomic max per dst.
// ---------------------------------------------------------------------------
__global__ void edge_score_kernel(const int* __restrict__ src, const int* __restrict__ dst,
                                  const unsigned char* __restrict__ mask,
                                  const float* __restrict__ xl, const float* __restrict__ xr,
                                  const float* __restrict__ avec,
                                  float* __restrict__ ew, float* __restrict__ segmax, int E) {
  int e = blockIdx.x * blockDim.x + threadIdx.x;
  if (e >= E) return;
  if (!mask[e]) return;
  const int s = src[e], d = dst[e];
  const float4* xls = reinterpret_cast<const float4*>(xl + (size_t)s * kH);
  const float4* xrs = reinterpret_cast<const float4*>(xr + (size_t)d * kH);
  const float4* av  = reinterpret_cast<const float4*>(avec);
  float acc = 0.0f;
#pragma unroll
  for (int i = 0; i < kH / 4; ++i) {
    float4 l = xls[i], r = xrs[i], a4 = av[i];
    float v;
    v = l.x + r.x; acc += a4.x * (v > 0.0f ? v : 0.2f * v);
    v = l.y + r.y; acc += a4.y * (v > 0.0f ? v : 0.2f * v);
    v = l.z + r.z; acc += a4.z * (v > 0.0f ? v : 0.2f * v);
    v = l.w + r.w; acc += a4.w * (v > 0.0f ? v : 0.2f * v);
  }
  ew[e] = acc;
  atomic_max_float(&segmax[d], acc);
}

// ---------------------------------------------------------------------------
// Edge pass 2: w = exp(e - segmax[dst]); accumulate denom per dst.
// Also prefetch xl[src] rows (global_prefetch_b8) to warm L2 for pass 3.
// ---------------------------------------------------------------------------
__global__ void edge_w_kernel(const int* __restrict__ src, const int* __restrict__ dst,
                              const unsigned char* __restrict__ mask,
                              float* __restrict__ ew, const float* __restrict__ segmax,
                              const float* __restrict__ xl,
                              float* __restrict__ denom, int E) {
  int e = blockIdx.x * blockDim.x + threadIdx.x;
  if (e >= E) return;
  float w = 0.0f;
  if (mask[e]) {
    const int d = dst[e];
    w = __expf(ew[e] - segmax[d]);
    atomicAdd(&denom[d], w);
    __builtin_prefetch(xl + (size_t)src[e] * kH, 0, 0);
  }
  ew[e] = w;
}

// ---------------------------------------------------------------------------
// Edge pass 3: agg[dst] += w * xl[src], thread per (edge, k).
// A wave covers 32 contiguous k of one edge -> coalesced gather + atomic scatter.
// ---------------------------------------------------------------------------
__global__ void edge_agg_kernel(const int* __restrict__ src, const int* __restrict__ dst,
                                const float* __restrict__ ew, const float* __restrict__ xl,
                                float* __restrict__ agg, int E) {
  size_t t = (size_t)blockIdx.x * blockDim.x + threadIdx.x;
  size_t tot = (size_t)E * kH;
  if (t >= tot) return;
  const int e = (int)(t >> 6);
  const int k = (int)(t & 63);
  const float w = ew[e];
  if (w == 0.0f) return;
  atomicAdd(&agg[(size_t)dst[e] * kH + k], w * xl[(size_t)src[e] * kH + k]);
}

// ---------------------------------------------------------------------------
// Conv finalize: accum += relu(agg / max(denom,1e-16) + b)
// ---------------------------------------------------------------------------
__global__ void conv_fin_kernel(const float* __restrict__ agg, const float* __restrict__ denom,
                                const float* __restrict__ bvec, float* __restrict__ accum,
                                size_t n64) {
  size_t i = (size_t)blockIdx.x * blockDim.x + threadIdx.x;
  if (i >= n64) return;
  const int node = (int)(i >> 6);
  const int k = (int)(i & 63);
  const float d = fmaxf(denom[node], 1e-16f);
  const float v = agg[i] / d + bvec[k];
  accum[i] += fmaxf(v, 0.0f);
}

// ---------------------------------------------------------------------------
// Row-wise L2 normalize (in place), thread per node.
// ---------------------------------------------------------------------------
__global__ void l2norm_kernel(float* __restrict__ x, int n) {
  int node = blockIdx.x * blockDim.x + threadIdx.x;
  if (node >= n) return;
  float4* row = reinterpret_cast<float4*>(x + (size_t)node * kH);
  float s = 0.0f;
#pragma unroll
  for (int i = 0; i < kH / 4; ++i) {
    float4 v = row[i];
    s += v.x * v.x + v.y * v.y + v.z * v.z + v.w * v.w;
  }
  const float inv = 1.0f / fmaxf(sqrtf(s), 1e-12f);
#pragma unroll
  for (int i = 0; i < kH / 4; ++i) {
    float4 v = row[i];
    v.x *= inv; v.y *= inv; v.z *= inv; v.w *= inv;
    row[i] = v;
  }
}

// ---------------------------------------------------------------------------
// Pooling: per (node,k) atomics into per-graph max/sum (+count once per node).
// ---------------------------------------------------------------------------
__global__ void pool_kernel(const float* __restrict__ x, const int* __restrict__ batch,
                            float* __restrict__ pmax, float* __restrict__ psum,
                            float* __restrict__ pcnt, int do_count, size_t n64) {
  size_t i = (size_t)blockIdx.x * blockDim.x + threadIdx.x;
  if (i >= n64) return;
  const int node = (int)(i >> 6);
  const int k = (int)(i & 63);
  const int g = batch[node];
  const float v = x[i];
  atomic_max_float(&pmax[(size_t)g * kH + k], v);
  atomicAdd(&psum[(size_t)g * kH + k], v);
  if (do_count && k == 0) atomicAdd(&pcnt[g], 1.0f);
}

// xc[g, 0:64]=gmp(x2), [64:128]=gap(x2), [128:192]=gmp(x3), [192:256]=gap(x3)
__global__ void pool_fin_kernel(const float* __restrict__ pmax2, const float* __restrict__ psum2,
                                const float* __restrict__ pmax3, const float* __restrict__ psum3,
                                const float* __restrict__ pcnt, float* __restrict__ xc) {
  int i = blockIdx.x * blockDim.x + threadIdx.x;
  if (i >= kG * kH) return;
  const int g = i >> 6;
  const int k = i & 63;
  const float c = fmaxf(pcnt[g], 1.0f);
  float m2 = pmax2[i]; if (!isfinite(m2)) m2 = 0.0f;
  float m3 = pmax3[i]; if (!isfinite(m3)) m3 = 0.0f;
  float* row = xc + (size_t)g * (4 * kH);
  row[k]            = m2;
  row[kH + k]       = psum2[i] / c;
  row[2 * kH + k]   = m3;
  row[3 * kH + k]   = psum3[i] / c;
}

// ---------------------------------------------------------------------------
// Head: logits = t2 @ lin3_W + lin3_b; out = log_softmax. Thread per graph.
// ---------------------------------------------------------------------------
__global__ void head_kernel(const float* __restrict__ t2, const float* __restrict__ W3,
                            const float* __restrict__ b3, float* __restrict__ out) {
  int g = blockIdx.x * blockDim.x + threadIdx.x;
  if (g >= kG) return;
  float logits[kC];
#pragma unroll
  for (int c = 0; c < kC; ++c) {
    float s = b3[c];
    for (int k = 0; k < kH; ++k) s += t2[(size_t)g * kH + k] * W3[(size_t)k * kC + c];
    logits[c] = s;
  }
  float m = logits[0];
#pragma unroll
  for (int c = 1; c < kC; ++c) m = fmaxf(m, logits[c]);
  float sum = 0.0f;
#pragma unroll
  for (int c = 0; c < kC; ++c) sum += __expf(logits[c] - m);
  const float lse = m + __logf(sum);
#pragma unroll
  for (int c = 0; c < kC; ++c) out[(size_t)g * kC + c] = logits[c] - lse;
}

// ---------------------------------------------------------------------------
// Host-side helpers
// ---------------------------------------------------------------------------
inline int cdiv(size_t a, size_t b) { return (int)((a + b - 1) / b); }

struct Scratch {
  float *h, *x1, *x2, *x3, *xl, *xr, *agg, *segmax, *denom, *ew;
  float *pmax2, *psum2, *pmax3, *psum3, *pcnt, *xc, *t1, *t2;
};

void run_gemm64(const float* A, const float* W, const float* bias, float* out,
                int M, int K, int relu, hipStream_t s) {
  gemm_wmma<4><<<cdiv((size_t)M, 128), 256, 0, s>>>(A, W, bias, out, M, K, relu);
}

void run_conv(const float* X, const float* Wl, const float* Wr,
              const float* avec, const float* bvec,
              const int* src, const int* dst, const unsigned char* mask,
              int E, float* accum, const Scratch& ws, hipStream_t s) {
  run_gemm64(X, Wl, nullptr, ws.xl, kN, kH, 0, s);
  run_gemm64(X, Wr, nullptr, ws.xr, kN, kH, 0, s);
  const size_t n64 = (size_t)kN * kH;
  fill_kernel<<<cdiv(kN, 256), 256, 0, s>>>(ws.segmax, kNeg, kN);
  fill_kernel<<<cdiv(kN, 256), 256, 0, s>>>(ws.denom, 0.0f, kN);
  fill_kernel<<<cdiv(n64, 256), 256, 0, s>>>(ws.agg, 0.0f, n64);
  edge_score_kernel<<<cdiv(E, 256), 256, 0, s>>>(src, dst, mask, ws.xl, ws.xr,
                                                 avec, ws.ew, ws.segmax, E);
  edge_w_kernel<<<cdiv(E, 256), 256, 0, s>>>(src, dst, mask, ws.ew, ws.segmax,
                                             ws.xl, ws.denom, E);
  edge_agg_kernel<<<cdiv((size_t)E * kH, 256), 256, 0, s>>>(src, dst, ws.ew, ws.xl,
                                                            ws.agg, E);
  conv_fin_kernel<<<cdiv(n64, 256), 256, 0, s>>>(ws.agg, ws.denom, bvec, accum, n64);
}

} // namespace

extern "C" void kernel_launch(void* const* d_in, const int* in_sizes, int n_in,
                              void* d_out, int out_size, void* d_ws, size_t ws_size,
                              hipStream_t stream) {
  (void)in_sizes; (void)n_in; (void)out_size; (void)ws_size;

  // ---- inputs (setup_inputs() insertion order; params flattened in order) ----
  const float* x          = (const float*)d_in[0];
  const int*   ei1        = (const int*)d_in[1];   // [2, E1]
  const int*   ei2        = (const int*)d_in[2];   // [2, E2]
  const int*   batch      = (const int*)d_in[3];
  const unsigned char* hom_mask = (const unsigned char*)d_in[4];
  const unsigned char* het_mask = (const unsigned char*)d_in[5];
  const unsigned char* hh_mask  = (const unsigned char*)d_in[6];
  const unsigned char* mm_mask  = (const unsigned char*)d_in[7];
  const unsigned char* tt_mask  = (const unsigned char*)d_in[8];
  const float* ft_W = (const float*)d_in[9];
  const float* ft_b = (const float*)d_in[10];
  // GAT param blocks: {Wl, Wr, a, b} starting at 11:
  // ghom0@11, ghet0@15, hom0@19, het0@23, mixed0@27, hom1@31, het1@35, mixed1@39
  auto gp = [&](int base, int off) { return (const float*)d_in[base + off]; };
  const float* lin1_W = (const float*)d_in[43];
  const float* lin1_b = (const float*)d_in[44];
  const float* lin2_W = (const float*)d_in[45];
  const float* lin2_b = (const float*)d_in[46];
  const float* lin3_W = (const float*)d_in[47];
  const float* lin3_b = (const float*)d_in[48];

  const int* s1 = ei1;           const int* d1 = ei1 + kE1;
  const int* s2 = ei2;           const int* d2 = ei2 + kE2;

  // ---- carve workspace (all offsets 256-float aligned) ----
  float* w = (float*)d_ws;
  size_t off = 0;
  auto take = [&](size_t n) { float* p = w + off; off += (n + 255) & ~(size_t)255; return p; };
  Scratch ws;
  const size_t n64 = (size_t)kN * kH;
  ws.h      = take(n64);
  ws.x1     = take(n64);
  ws.x2     = take(n64);
  ws.x3     = take(n64);
  ws.xl     = take(n64);
  ws.xr     = take(n64);
  ws.agg    = take(n64);
  ws.segmax = take(kN);
  ws.denom  = take(kN);
  ws.ew     = take(kE2);
  ws.pmax2  = take((size_t)kG * kH);
  ws.psum2  = take((size_t)kG * kH);
  ws.pmax3  = take((size_t)kG * kH);
  ws.psum3  = take((size_t)kG * kH);
  ws.pcnt   = take(kG);
  ws.xc     = take((size_t)kG * 4 * kH);
  ws.t1     = take((size_t)kG * 2 * kH);
  ws.t2     = take((size_t)kG * kH);

  // ---- feature transform: h = x @ ft_W + ft_b ----
  run_gemm64(x, ft_W, ft_b, ws.h, kN, kFeat, 0, stream);

  // ---- layer 0 (1-hop, two views): x1 = h + relu(gat) + relu(gat); l2norm ----
  copy_kernel<<<cdiv(n64, 256), 256, 0, stream>>>(ws.h, ws.x1, n64);
  run_conv(ws.h, gp(11,0), gp(11,1), gp(11,2), gp(11,3), s1, d1, hom_mask, kE1, ws.x1, ws, stream);
  run_conv(ws.h, gp(15,0), gp(15,1), gp(15,2), gp(15,3), s1, d1, het_mask, kE1, ws.x1, ws, stream);
  l2norm_kernel<<<cdiv(kN, 256), 256, 0, stream>>>(ws.x1, kN);

  // ---- layer 1 (2-hop, three views) -> x2 ----
  copy_kernel<<<cdiv(n64, 256), 256, 0, stream>>>(ws.x1, ws.x2, n64);
  run_conv(ws.x1, gp(19,0), gp(19,1), gp(19,2), gp(19,3), s2, d2, hh_mask, kE2, ws.x2, ws, stream);
  run_conv(ws.x1, gp(23,0), gp(23,1), gp(23,2), gp(23,3), s2, d2, tt_mask, kE2, ws.x2, ws, stream);
  run_conv(ws.x1, gp(27,0), gp(27,1), gp(27,2), gp(27,3), s2, d2, mm_mask, kE2, ws.x2, ws, stream);
  l2norm_kernel<<<cdiv(kN, 256), 256, 0, stream>>>(ws.x2, kN);

  // ---- pooling of x2 ----
  const size_t g64 = (size_t)kG * kH;
  fill_kernel<<<cdiv(g64, 256), 256, 0, stream>>>(ws.pmax2, -__builtin_huge_valf(), g64);
  fill_kernel<<<cdiv(g64, 256), 256, 0, stream>>>(ws.psum2, 0.0f, g64);
  fill_kernel<<<cdiv((size_t)kG, 256), 256, 0, stream>>>(ws.pcnt, 0.0f, kG);
  pool_kernel<<<cdiv(n64, 256), 256, 0, stream>>>(ws.x2, batch, ws.pmax2, ws.psum2,
                                                  ws.pcnt, 1, n64);

  // ---- layer 2 (2-hop, three views) -> x3 ----
  copy_kernel<<<cdiv(n64, 256), 256, 0, stream>>>(ws.x2, ws.x3, n64);
  run_conv(ws.x2, gp(31,0), gp(31,1), gp(31,2), gp(31,3), s2, d2, hh_mask, kE2, ws.x3, ws, stream);
  run_conv(ws.x2, gp(35,0), gp(35,1), gp(35,2), gp(35,3), s2, d2, tt_mask, kE2, ws.x3, ws, stream);
  run_conv(ws.x2, gp(39,0), gp(39,1), gp(39,2), gp(39,3), s2, d2, mm_mask, kE2, ws.x3, ws, stream);
  l2norm_kernel<<<cdiv(kN, 256), 256, 0, stream>>>(ws.x3, kN);

  // ---- pooling of x3 (counts already computed) ----
  fill_kernel<<<cdiv(g64, 256), 256, 0, stream>>>(ws.pmax3, -__builtin_huge_valf(), g64);
  fill_kernel<<<cdiv(g64, 256), 256, 0, stream>>>(ws.psum3, 0.0f, g64);
  pool_kernel<<<cdiv(n64, 256), 256, 0, stream>>>(ws.x3, batch, ws.pmax3, ws.psum3,
                                                  ws.pcnt, 0, n64);
  pool_fin_kernel<<<cdiv((size_t)kG * kH, 256), 256, 0, stream>>>(
      ws.pmax2, ws.psum2, ws.pmax3, ws.psum3, ws.pcnt, ws.xc);

  // ---- head MLP: lin1 (WMMA, relu) -> lin2 (WMMA, relu) -> lin3 + log_softmax
  gemm_wmma<8><<<cdiv((size_t)kG, 128), 256, 0, stream>>>(ws.xc, lin1_W, lin1_b,
                                                          ws.t1, kG, 4 * kH, 1);
  gemm_wmma<4><<<cdiv((size_t)kG, 128), 256, 0, stream>>>(ws.t1, lin2_W, lin2_b,
                                                          ws.t2, kG, 2 * kH, 1);
  head_kernel<<<cdiv((size_t)kG, 256), 256, 0, stream>>>(ws.t2, lin3_W, lin3_b,
                                                         (float*)d_out);
}
